// Attention_33887291965472
// MI455X (gfx1250) — matrix operand order repaired and checked by reading, not compiled, not run on hardware
//
#include <hip/hip_runtime.h>

// ---------------------------------------------------------------------------
// GPT-style attention block for MI455X (gfx1250), wave32 + WMMA f16.
//   B=2, S=2048, D_MODEL=1024, H=16, D_HEAD=64, rotary_dim=64, scale=8.
// ~90 GFLOP vs ~50MB traffic -> compute bound -> matrix pipeline everywhere.
// All B operands (weights, K^T, V) are stored in WMMA *fragment-major* order
// ([block of 32 lanes x 16 halves]) so every fragment load is a contiguous
// 32B-per-lane read (global_load_b128 / ds_load_b128) with no VALU packing.
// Workspace: 48 MiB of d_ws.
// ---------------------------------------------------------------------------

#define DMODEL 1024
#define NHEADS 16
#define DHEAD  64
#define SEQ    2048
#define BATCH  2

typedef _Float16 half_t;
typedef __attribute__((ext_vector_type(16))) _Float16 v16h;
typedef __attribute__((ext_vector_type(8)))  _Float16 v8h;
typedef __attribute__((ext_vector_type(8)))  float    v8f;

__device__ __forceinline__ v8f wmma16(v16h a, v16h b, v8f c) {
  // (neg_a, A, neg_b, B, c_mod, C, reuse_a, reuse_b)
  return __builtin_amdgcn_wmma_f32_16x16x32_f16(false, a, false, b, (short)0, c,
                                                false, false);
}

__device__ __forceinline__ v16h cat8(v8h lo, v8h hi) {
  return __builtin_shufflevector(lo, hi, 0, 1, 2, 3, 4, 5, 6, 7, 8, 9, 10, 11,
                                 12, 13, 14, 15);
}

// A-matrix 16x32 f16 order used here (k = 16h + s, same as B): lane<16 -> K {0..15},
// lane>=16 -> K {16..31}: two contiguous 8-half runs at 2*k8 and 2*k8+8.
// B-matrix 32x16 f16 layout: lane (n=l&15, koff=(l&16)) holds B[koff+t][n].

// ---------------------------------------------------------------------------
__global__ void cvt_f16(const float* __restrict__ s, half_t* __restrict__ d,
                        int n) {
  int i = blockIdx.x * blockDim.x + threadIdx.x;
  if (i < n) d[i] = (half_t)s[i];
}

// f32 [K][N] row-major -> f16 fragment-major:
// dst[((kc*(N/16)+nt)*32 + lane)*16 + t] = src[(kc*32 + (lane&16) + t)*N
//                                              + nt*16 + (lane&15)]
__global__ void swizzle_b(const float* __restrict__ s, half_t* __restrict__ d,
                          int K, int N) {
  int e = blockIdx.x * blockDim.x + threadIdx.x;
  if (e >= K * N) return;
  int t    = e & 15;
  int lane = (e >> 4) & 31;
  int blk  = e >> 9;
  int NT   = N >> 4;
  int nt   = blk % NT;
  int kc   = blk / NT;
  int row  = kc * 32 + (lane & 16) + t;
  int col  = nt * 16 + (lane & 15);
  d[e] = (half_t)s[row * N + col];
}

// ---------------------------------------------------------------------------
// QKV projection + bias + RoPE. Grid: (S/16, H, B), 128 threads (4 waves).
// Wave w computes the 16-column slice [w*16, w*16+16) of Q, K, V.
// Outputs: Q row-major [b,h,s,d] (A operand), K & V fragment-swizzled panels.
// ---------------------------------------------------------------------------
__global__ __launch_bounds__(128) void qkv_rope_kernel(
    const half_t* __restrict__ xh,
    const half_t* __restrict__ wq, const half_t* __restrict__ wk,
    const half_t* __restrict__ wv,
    const float* __restrict__ bq, const float* __restrict__ bk,
    const float* __restrict__ bv,
    half_t* __restrict__ Qh, half_t* __restrict__ Ksw, half_t* __restrict__ Vsw)
{
  __shared__ __align__(32) half_t xs[16 * DMODEL];  // 32 KB shared A tile
  __shared__ float qs[16 * DHEAD];                  // f32 staging for RoPE
  __shared__ float ks[16 * DHEAD];
  __shared__ float vs[16 * DHEAD];

  const int sblk  = blockIdx.x;
  const int h     = blockIdx.y;
  const int b     = blockIdx.z;
  const int tid   = threadIdx.x;
  const int lane  = tid & 31;
  const int wid   = tid >> 5;
  const int bh    = b * NHEADS + h;
  const int sbase = sblk * 16;
  const int ln    = lane & 15;
  const int k8    = (lane < 16) ? 0 : 8;
  const int hrow  = (lane < 16) ? 0 : 8;

  // cooperative load of x tile [16 x 1024] (rows contiguous), 16B per thread
  {
    const uint4* src = (const uint4*)(xh + ((size_t)b * SEQ + sbase) * DMODEL);
    uint4* dst = (uint4*)xs;
#pragma unroll
    for (int p = tid; p < 16 * DMODEL / 8; p += 128) dst[p] = src[p];
  }
  __syncthreads();

  const int ncol = wid * 16;
  v8f aq = {}, ak = {}, av = {};
#pragma unroll 2
  for (int kk = 0; kk < DMODEL / 32; ++kk) {
    // A frag: two contiguous 8-half runs from LDS -> ds_load_b128 x2
    const half_t* ap = &xs[ln * DMODEL + kk * 32];
    v16h afrag = cat8(*(const v8h*)(ap + 2 * k8), *(const v8h*)(ap + 2 * k8 + 8));
    // B frags: fragment-major weights -> 32 contiguous bytes per lane
    size_t boff = ((size_t)((h * 32 + kk) * 4 + wid) * 32 + lane) * 16;
    v16h bql = *(const v16h*)(wq + boff);
    v16h bkl = *(const v16h*)(wk + boff);
    v16h bvl = *(const v16h*)(wv + boff);
    aq = wmma16(afrag, bql, aq);
    ak = wmma16(afrag, bkl, ak);
    av = wmma16(afrag, bvl, av);
  }

  // stage C tiles (f32) + bias for the swizzled epilogues
#pragma unroll
  for (int r = 0; r < 8; ++r) {
    int m = r + hrow;
    int d = ncol + ln;
    qs[m * DHEAD + d] = aq[r] + bq[h * DHEAD + d];
    ks[m * DHEAD + d] = ak[r] + bk[h * DHEAD + d];
    vs[m * DHEAD + d] = av[r] + bv[h * DHEAD + d];
  }
  __syncthreads();

  // --- Q: RoPE, row-major [b,h,s,d] (consumed as WMMA-A) ---
  for (int idx = tid; idx < 16 * DHEAD; idx += 128) {
    int row = idx >> 6;
    int d   = idx & 63;
    int dm  = d & 31;
    float inv = __expf(-(float)dm * 0.2878231366f);  // 10000^(-dm/32)
    float sn, cs;
    __sincosf((float)(sbase + row) * inv, &sn, &cs);
    float qv = qs[row * DHEAD + d];
    float qf = (d < 32) ? -qs[row * DHEAD + d + 32] : qs[row * DHEAD + d - 32];
    Qh[((size_t)bh * SEQ + sbase + row) * DHEAD + d] = (half_t)(qv * cs + qf * sn);
  }

  // --- K: RoPE, fragment-swizzled panel Ksw[bh][s16][kkd][lane][t] ---
  // frag element (l,t) of block (s16,kkd) = K[key=s16*16+(l&15)][d=kkd*32+(l&16)+t]
  for (int idx = tid; idx < 16 * DHEAD; idx += 128) {
    int kkd = idx >> 9;
    int rem = idx & 511;
    int l   = rem >> 4;
    int t   = rem & 15;
    int row = l & 15;
    int d   = kkd * 32 + (l & 16) + t;
    int dm  = d & 31;
    float inv = __expf(-(float)dm * 0.2878231366f);
    float sn, cs;
    __sincosf((float)(sbase + row) * inv, &sn, &cs);
    float kv = ks[row * DHEAD + d];
    float kf = (d < 32) ? -ks[row * DHEAD + d + 32] : ks[row * DHEAD + d - 32];
    Ksw[((size_t)(bh * (SEQ / 16) + sblk) * 2 + kkd) * 512 + rem] =
        (half_t)(kv * cs + kf * sn);
  }

  // --- V: fragment-swizzled panel Vsw[bh][kt32][t2][lane][t] ---
  // frag element (l,t) of block (kt,t2) = V[key=kt*32+(l&16)+t][d=t2*16+(l&15)]
  {
    const int kt   = sblk >> 1;
    const int half = sblk & 1;  // this block provides lanes half*16..+15
    for (int idx = tid; idx < 16 * DHEAD; idx += 128) {
      int t2 = idx >> 8;
      int rem = idx & 255;
      int lq = rem >> 4;
      int t  = rem & 15;
      float val = vs[t * DHEAD + t2 * 16 + lq];
      Vsw[((size_t)(bh * (SEQ / 32) + kt) * 4 + t2) * 512 +
          (lq + half * 16) * 16 + t] = (half_t)val;
    }
  }
}

// ---------------------------------------------------------------------------
// Flash attention. Grid: (S/64, H, B), 128 threads = 4 waves; wave w owns
// q rows [qblk*64 + w*16, +16). 32-key tiles staged in LDS (pre-swizzled).
// ---------------------------------------------------------------------------
__global__ __launch_bounds__(128) void attn_kernel(
    const half_t* __restrict__ Qh, const half_t* __restrict__ Ksw,
    const half_t* __restrict__ Vsw, half_t* __restrict__ Zh)
{
  __shared__ __align__(32) half_t kls[2048];      // 4 K frag blocks (4 KB)
  __shared__ __align__(32) half_t vls[2048];      // 4 V frag blocks (4 KB)
  __shared__ __align__(32) half_t psh[4][16 * 32];// per-wave P staging

  const int qblk  = blockIdx.x;
  const int h     = blockIdx.y;
  const int b     = blockIdx.z;
  const int tid   = threadIdx.x;
  const int lane  = tid & 31;
  const int w     = tid >> 5;
  const int bh    = b * NHEADS + h;
  const int qbase = qblk * 64 + w * 16;
  const int ln    = lane & 15;
  const int k8    = (lane < 16) ? 0 : 8;
  const int hrow  = (lane < 16) ? 0 : 8;

  // Q A-fragments (16 rows x 64 dh -> two 16x32 frags), loaded once, b128s
  const half_t* Qp = Qh + ((size_t)bh * SEQ + qbase) * DHEAD + ln * DHEAD;
  v16h qa0 = cat8(*(const v8h*)(Qp + 2 * k8), *(const v8h*)(Qp + 2 * k8 + 8));
  v16h qa1 = cat8(*(const v8h*)(Qp + 32 + 2 * k8), *(const v8h*)(Qp + 40 + 2 * k8));

  v8f o[4] = {};
  float mrow[8], lrow[8];
#pragma unroll
  for (int r = 0; r < 8; ++r) { mrow[r] = -1e30f; lrow[r] = 0.0f; }

  const half_t* Kp = Ksw + (size_t)bh * (SEQ / 16) * 2 * 512;
  const half_t* Vp = Vsw + (size_t)bh * (SEQ / 32) * 4 * 512;

  const int nkt = qblk * 2 + 2;  // 32-key tiles needed by last q row of block
  for (int kt = 0; kt < nkt; ++kt) {
    const int ktbase = kt * 32;
    __syncthreads();  // WAR guard on LDS tiles
    {
      // both panels are contiguous 4 KB runs -> pure uint4 copies
      const uint4* ksrc = (const uint4*)(Kp + (size_t)kt * 2048);
      const uint4* vsrc = (const uint4*)(Vp + (size_t)kt * 2048);
      uint4* kdst = (uint4*)kls;
      uint4* vdst = (uint4*)vls;
#pragma unroll
      for (int p = tid; p < 256; p += 128) {
        kdst[p] = ksrc[p];
        vdst[p] = vsrc[p];
      }
    }
    if (kt + 1 < nkt)  // pull next K panel toward L2 (global_prefetch_b8)
      __builtin_prefetch(Kp + (size_t)(kt + 1) * 2048, 0, 1);
    __syncthreads();

    if (ktbase > qbase + 15) continue;  // tile fully masked for this wave

    // scores: S(16x32) = Q(16x64) x K^T(64x32); frag block (sub,kk) at
    // kls + (sub*2+kk)*512, 32 contiguous bytes per lane -> ds_load_b128 x2
    v8f s0 = {}, s1 = {};
#pragma unroll
    for (int kk = 0; kk < 2; ++kk) {
      v16h bk0 = *(const v16h*)(&kls[(0 * 2 + kk) * 512 + lane * 16]);
      v16h bk1 = *(const v16h*)(&kls[(1 * 2 + kk) * 512 + lane * 16]);
      v16h aq = (kk == 0) ? qa0 : qa1;
      s0 = wmma16(aq, bk0, s0);
      s1 = wmma16(aq, bk1, s1);
    }

    // online softmax; row elements live across one 16-lane half per VGPR
    float corr[8];
#pragma unroll
    for (int r = 0; r < 8; ++r) {
      int   q_abs = qbase + r + hrow;
      int   k0    = ktbase + ln;
      float x0 = (k0 <= q_abs)      ? s0[r] * 0.125f : -1e30f;
      float x1 = (k0 + 16 <= q_abs) ? s1[r] * 0.125f : -1e30f;
      float mx = fmaxf(x0, x1);
      mx = fmaxf(mx, __shfl_xor(mx, 1));
      mx = fmaxf(mx, __shfl_xor(mx, 2));
      mx = fmaxf(mx, __shfl_xor(mx, 4));
      mx = fmaxf(mx, __shfl_xor(mx, 8));
      float mnew = fmaxf(mrow[r], mx);
      float c    = __expf(mrow[r] - mnew);
      float p0   = __expf(x0 - mnew);
      float p1   = __expf(x1 - mnew);
      float rs   = p0 + p1;
      rs += __shfl_xor(rs, 1);
      rs += __shfl_xor(rs, 2);
      rs += __shfl_xor(rs, 4);
      rs += __shfl_xor(rs, 8);
      lrow[r] = lrow[r] * c + rs;
      mrow[r] = mnew;
      corr[r] = c;
      int m = r + hrow;  // C-layout -> P staging (f16)
      psh[w][m * 32 + ln]      = (half_t)p0;
      psh[w][m * 32 + 16 + ln] = (half_t)p1;
    }
#pragma unroll
    for (int t2 = 0; t2 < 4; ++t2)
#pragma unroll
      for (int r = 0; r < 8; ++r) o[t2][r] *= corr[r];

    // wave-local LDS relayout C->A; force DS completion before re-read
    asm volatile("s_wait_dscnt 0" ::: "memory");
    const half_t* pp = &psh[w][ln * 32];
    v16h pa = cat8(*(const v8h*)(pp + 2 * k8), *(const v8h*)(pp + 2 * k8 + 8));

    // O += P(16x32) x V(32x64); V frag block t2 at vls + t2*512
#pragma unroll
    for (int t2 = 0; t2 < 4; ++t2) {
      v16h bv = *(const v16h*)(&vls[t2 * 512 + lane * 16]);
      o[t2] = wmma16(pa, bv, o[t2]);
    }
  }

  // normalize and write Z[b, q, h*64 + d] (f16, row-major A operand for proj)
#pragma unroll
  for (int r = 0; r < 8; ++r) {
    float invl = 1.0f / lrow[r];
    int   m    = r + hrow;
    size_t base = ((size_t)b * SEQ + qbase + m) * DMODEL + h * DHEAD + ln;
#pragma unroll
    for (int t2 = 0; t2 < 4; ++t2)
      Zh[base + t2 * 16] = (half_t)(o[t2][r] * invl);
  }
}

// ---------------------------------------------------------------------------
// Output projection: out = Z[4096x1024] x W_O[1024x1024] + b_O (f32 out).
// Grid: (B*S/16, DMODEL/64), 128 threads (4 waves, 16 cols each).
// ---------------------------------------------------------------------------
__global__ __launch_bounds__(128) void proj_kernel(
    const half_t* __restrict__ Zh, const half_t* __restrict__ Wo,
    const float* __restrict__ bo, float* __restrict__ out)
{
  __shared__ __align__(32) half_t zs[16 * DMODEL];
  const int mblk = blockIdx.x;
  const int nblk = blockIdx.y;
  const int tid  = threadIdx.x;
  const int lane = tid & 31;
  const int w    = tid >> 5;
  const int ln   = lane & 15;
  const int k8   = (lane < 16) ? 0 : 8;
  const int hrow = (lane < 16) ? 0 : 8;
  const int ncol = nblk * 64 + w * 16;

  {
    const uint4* src = (const uint4*)(Zh + (size_t)mblk * 16 * DMODEL);
    uint4* dst = (uint4*)zs;
#pragma unroll
    for (int p = tid; p < 16 * DMODEL / 8; p += 128) dst[p] = src[p];
  }
  __syncthreads();

  v8f acc = {};
#pragma unroll 2
  for (int kk = 0; kk < DMODEL / 32; ++kk) {
    const half_t* ap = &zs[ln * DMODEL + kk * 32];
    v16h a = cat8(*(const v8h*)(ap + 2 * k8), *(const v8h*)(ap + 2 * k8 + 8));
    v16h bf = *(const v16h*)(
        Wo + ((size_t)(kk * 64 + nblk * 4 + w) * 32 + lane) * 16);
    acc = wmma16(a, bf, acc);
  }
#pragma unroll
  for (int r = 0; r < 8; ++r) {
    int m = r + hrow;
    out[((size_t)mblk * 16 + m) * DMODEL + ncol + ln] = acc[r] + bo[ncol + ln];
  }
}

// ---------------------------------------------------------------------------
extern "C" void kernel_launch(void* const* d_in, const int* in_sizes, int n_in,
                              void* d_out, int out_size, void* d_ws,
                              size_t ws_size, hipStream_t stream) {
  const float* x  = (const float*)d_in[0];
  const float* WQ = (const float*)d_in[1];
  const float* bQ = (const float*)d_in[2];
  const float* WK = (const float*)d_in[3];
  const float* bK = (const float*)d_in[4];
  const float* WV = (const float*)d_in[5];
  const float* bV = (const float*)d_in[6];
  const float* WO = (const float*)d_in[7];
  const float* bO = (const float*)d_in[8];
  float* out = (float*)d_out;

  // workspace layout (48 MiB total)
  char* ws = (char*)d_ws;
  half_t* xh  = (half_t*)(ws + 0);                         // 8 MiB
  half_t* wq  = (half_t*)(ws + (size_t) 8 * 1024 * 1024);  // 2 MiB (swizzled)
  half_t* wk  = (half_t*)(ws + (size_t)10 * 1024 * 1024);  // 2 MiB (swizzled)
  half_t* wv  = (half_t*)(ws + (size_t)12 * 1024 * 1024);  // 2 MiB (swizzled)
  half_t* wo  = (half_t*)(ws + (size_t)14 * 1024 * 1024);  // 2 MiB (swizzled)
  half_t* Qh  = (half_t*)(ws + (size_t)16 * 1024 * 1024);  // 8 MiB row-major
  half_t* Ksw = (half_t*)(ws + (size_t)24 * 1024 * 1024);  // 8 MiB frag-major
  half_t* Vsw = (half_t*)(ws + (size_t)32 * 1024 * 1024);  // 8 MiB frag-major
  half_t* Zh  = (half_t*)(ws + (size_t)40 * 1024 * 1024);  // 8 MiB row-major

  const int nx = BATCH * SEQ * DMODEL;
  const int nw = NHEADS * DMODEL * DHEAD;
  cvt_f16<<<(nx + 255) / 256, 256, 0, stream>>>(x, xh, nx);
  // QKV weights: heads stack along K (16*1024 rows x 64 cols)
  swizzle_b<<<(nw + 255) / 256, 256, 0, stream>>>(WQ, wq, NHEADS * DMODEL, DHEAD);
  swizzle_b<<<(nw + 255) / 256, 256, 0, stream>>>(WK, wk, NHEADS * DMODEL, DHEAD);
  swizzle_b<<<(nw + 255) / 256, 256, 0, stream>>>(WV, wv, NHEADS * DMODEL, DHEAD);
  swizzle_b<<<(nw + 255) / 256, 256, 0, stream>>>(WO, wo, DMODEL, DMODEL);

  qkv_rope_kernel<<<dim3(SEQ / 16, NHEADS, BATCH), 128, 0, stream>>>(
      xh, wq, wk, wv, bQ, bK, bV, Qh, Ksw, Vsw);

  attn_kernel<<<dim3(SEQ / 64, NHEADS, BATCH), 128, 0, stream>>>(Qh, Ksw, Vsw,
                                                                 Zh);

  proj_kernel<<<dim3(BATCH * SEQ / 16, DMODEL / 64), 128, 0, stream>>>(
      Zh, wo, bO, out);
}
